// LocusModule_52905407152558
// MI455X (gfx1250) — matrix operand order repaired and checked by reading, not compiled, not run on hardware
//
#include <hip/hip_runtime.h>
#include <hip/hip_bf16.h>

#define B_   4096
#define L_   20
#define POS  (B_ * L_)          // 81920 positions per branch
#define EPS_ 1e-5f
#define TEMP_ 0.03f

typedef __attribute__((ext_vector_type(16))) _Float16 v16h;
typedef __attribute__((ext_vector_type(8)))  float    v8f;
typedef __attribute__((ext_vector_type(4)))  unsigned v4u;

union Frag16 { v16h v; v4u q[2]; };

// ---------------- stateless hash RNG (structural stand-in for jax threefry) ---
__device__ __forceinline__ unsigned hash3(unsigned a, unsigned b, unsigned c) {
    unsigned h = a * 0x9E3779B9u ^ b * 0x85EBCA6Bu ^ c * 0xC2B2AE35u;
    h ^= h >> 16; h *= 0x7FEB352Du;
    h ^= h >> 15; h *= 0x846CA68Bu;
    h ^= h >> 16;
    return h;
}
__device__ __forceinline__ float u01(unsigned h) { return (h >> 8) * (1.0f / 16777216.0f); }

// ---------------- monotone float<->uint encoding for atomicMax --------------
__device__ __forceinline__ unsigned fenc(float f) {
    unsigned u = __float_as_uint(f);
    return (u & 0x80000000u) ? ~u : (u | 0x80000000u);
}
__device__ __forceinline__ float fdec(unsigned k) {
    unsigned u = (k & 0x80000000u) ? (k & 0x7FFFFFFFu) : ~k;
    return __uint_as_float(u);
}

// ---------------- augmentation + bn_time batch stats ------------------------
__global__ void aug_stats_kernel(const float* __restrict__ cid, const float* __restrict__ t,
                                 float* c0, float* t0, float* c1, float* t1,
                                 float* tstat /* [2][40]: sum[20], sumsq[20] per branch */) {
    __shared__ float s[40];
    int tid = threadIdx.x;
    if (tid < 40) s[tid] = 0.f;
    __syncthreads();
    int idx = blockIdx.x * blockDim.x + tid;       // exactly 2*POS threads
    int br  = idx / POS;
    int p   = idx % POS;
    int l   = p % L_;
    float u  = u01(hash3(br, p, 0));
    float rc = (float)(hash3(br, p, 1) % 51u);
    float ta = -50.f + 100.f * u01(hash3(br, p, 2));
    float tb = -50.f + 100.f * u01(hash3(br, p, 3));
    float tc = -5.f  + 10.f  * u01(hash3(br, p, 4));
    float c = cid[p], tt = t[p];
    float nc = (u < 0.4f) ? rc : c;
    float dt = (u < 0.2f) ? ta : (u < 0.4f) ? 0.f : (u < 0.6f) ? tb : tc;
    float nt = tt + dt;
    if (br == 0) { c0[p] = nc; t0[p] = nt; } else { c1[p] = nc; t1[p] = nt; }
    atomicAdd(&s[l], nt);
    atomicAdd(&s[20 + l], nt * nt);
    __syncthreads();
    if (tid < 40) atomicAdd(&tstat[br * 40 + tid], s[tid]);
}

// ---------------- small helpers ---------------------------------------------
__global__ void f2h_kernel(const float* __restrict__ x, _Float16* __restrict__ y, int n) {
    int i = blockIdx.x * blockDim.x + threadIdx.x;
    if (i < n) y[i] = (_Float16)x[i];
}

// stage1: x=[cid, bn_time(t)] (2 ch) -> y1[p][64] = W1 x + b1 (pre-BN)
__global__ void stage1_kernel(const float* __restrict__ c, const float* __restrict__ t,
                              const float* __restrict__ tstat, int br,
                              const float* __restrict__ W1, const float* __restrict__ b1,
                              const float* __restrict__ gt, const float* __restrict__ bt,
                              float* __restrict__ y1) {
    int idx = blockIdx.x * blockDim.x + threadIdx.x;   // POS*64
    int p = idx >> 6, o = idx & 63;
    int l = p % L_;
    float m  = tstat[br * 40 + l] * (1.f / B_);
    float v  = tstat[br * 40 + 20 + l] * (1.f / B_) - m * m;
    float rs = rsqrtf(v + EPS_);
    float tn = gt[l] * (t[p] - m) * rs + bt[l];
    y1[idx] = W1[o * 2] * c[p] + W1[o * 2 + 1] * tn + b1[o];
}

// per-column sum/sumsq of a [M,N] fp32 matrix
__global__ void colstats_kernel(const float* __restrict__ x, float* sum, float* sumsq,
                                long M, int N) {
    extern __shared__ float s[];   // 2*N floats
    for (int i = threadIdx.x; i < 2 * N; i += blockDim.x) s[i] = 0.f;
    __syncthreads();
    long total = M * (long)N;
    for (long i = blockIdx.x * (long)blockDim.x + threadIdx.x; i < total;
         i += (long)gridDim.x * blockDim.x) {
        int col = (int)(i % N);
        float v = x[i];
        atomicAdd(&s[col], v);
        atomicAdd(&s[N + col], v * v);
    }
    __syncthreads();
    for (int i = threadIdx.x; i < N; i += blockDim.x) {
        atomicAdd(&sum[i], s[i]);
        atomicAdd(&sumsq[i], s[N + i]);
    }
}

// fold BN into per-channel affine: bn(y) = a*y + c
__global__ void bnfinal_kernel(const float* sum, const float* sumsq,
                               const float* g, const float* be,
                               float* bnA, float* bnC, int N, float invCnt) {
    int n = blockIdx.x * blockDim.x + threadIdx.x;
    if (n >= N) return;
    float m  = sum[n] * invCnt;
    float v  = sumsq[n] * invCnt - m * m;
    float a  = g[n] * rsqrtf(v + EPS_);
    bnA[n] = a;
    bnC[n] = be[n] - m * a;
}

// h = fp16(relu(a*y + c)) elementwise over [M,N]
__global__ void bnrelu_h_kernel(const float* __restrict__ y, const float* __restrict__ bnA,
                                const float* __restrict__ bnC, _Float16* __restrict__ out,
                                int N, long total) {
    long i = blockIdx.x * (long)blockDim.x + threadIdx.x;
    if (i >= total) return;
    int o = (int)(i % N);
    out[i] = (_Float16)fmaxf(0.f, bnA[o] * y[i] + bnC[o]);
}

// ---------------- WMMA GEMM: C = A[M,K] * Bt[N,K]^T, fused epilogues --------
// Register-blocked: each wave computes a 64x32 macro-tile = 4x2 grid of 16x16
// WMMA tiles (8 wmma per 6 fragment loads per 32-K step).
// MODE 0: acc+bias -> per-channel sum/sumsq atomics (no store)
// MODE 1: fp16(relu(bnA*(acc+bias)+bnC)) stored row-major [M,N]
// MODE 2: bnA*(acc+bias)+bnC -> atomicMax over L into [M/L, N] (encoded uint)
// MODE 3: alpha*acc stored fp32 row-major [M,N]
template <int MODE>
__global__ void wmma_gemm_kernel(const _Float16* __restrict__ A, const _Float16* __restrict__ Bt,
                                 int M, int N, int K,
                                 const float* __restrict__ bias,
                                 const float* __restrict__ bnA, const float* __restrict__ bnC,
                                 float* sum, float* sumsq,
                                 _Float16* outH, float* outF, unsigned* outMax,
                                 float alpha) {
    constexpr int TM = 4, TN = 2;                 // 16x16 tiles per wave
    int lane = threadIdx.x & 31;
    int wave = blockIdx.x * (blockDim.x >> 5) + (threadIdx.x >> 5);
    int gN = N >> 5;                              // 32-wide macro-tiles across N
    int gM = M >> 6;                              // 64-tall macro-tiles down M
    if (wave >= gM * gN) return;
    int row0 = (wave / gN) << 6;
    int col0 = (wave % gN) << 5;
    int mn = lane & 15, hi = lane >> 4;

    // per-lane fragment base pointers (A 16x32 / B 32x16 f16 layouts, ISA 7.12.2)
    const _Float16* aRow[TM];
    const _Float16* bRow[TN];
#pragma unroll
    for (int i = 0; i < TM; ++i) aRow[i] = A + (long)(row0 + i * 16 + mn) * K + 8 * hi;
#pragma unroll
    for (int j = 0; j < TN; ++j) bRow[j] = Bt + (long)(col0 + j * 16 + mn) * K + 8 * hi;

    v8f acc[TM][TN] = {};
    for (int k0 = 0; k0 < K; k0 += 32) {
        Frag16 fa[TM], fb[TN];
#pragma unroll
        for (int i = 0; i < TM; ++i) {
            fa[i].q[0] = *(const v4u*)(aRow[i] + k0);
            fa[i].q[1] = *(const v4u*)(aRow[i] + k0 + 16);
            __builtin_prefetch(aRow[i] + k0 + 32, 0, 3);   // speculative, drop-safe
        }
#pragma unroll
        for (int j = 0; j < TN; ++j) {
            fb[j].q[0] = *(const v4u*)(bRow[j] + k0);
            fb[j].q[1] = *(const v4u*)(bRow[j] + k0 + 16);
            __builtin_prefetch(bRow[j] + k0 + 32, 0, 3);
        }
#pragma unroll
        for (int i = 0; i < TM; ++i)
#pragma unroll
            for (int j = 0; j < TN; ++j)
                acc[i][j] = __builtin_amdgcn_wmma_f32_16x16x32_f16(
                    false, fa[i].v, false, fb[j].v, (short)0, acc[i][j], false, false);
    }

#pragma unroll
    for (int i = 0; i < TM; ++i) {
        int tr0 = row0 + i * 16;
#pragma unroll
        for (int j = 0; j < TN; ++j) {
            int col = col0 + j * 16 + mn;
            v8f a8 = acc[i][j];
            if (MODE == 0) {
                float bs = bias ? bias[col] : 0.f;
                float s = 0.f, s2 = 0.f;
#pragma unroll
                for (int r = 0; r < 8; ++r) { float y = a8[r] + bs; s += y; s2 += y * y; }
                s  += __shfl_xor(s, 16, 32);
                s2 += __shfl_xor(s2, 16, 32);
                if (lane < 16) { atomicAdd(&sum[col], s); atomicAdd(&sumsq[col], s2); }
            } else if (MODE == 1) {
                float bs = bias[col], a = bnA[col], c = bnC[col];
#pragma unroll
                for (int r = 0; r < 8; ++r) {
                    int row = tr0 + r + 8 * hi;
                    outH[(long)row * N + col] = (_Float16)fmaxf(0.f, a * (a8[r] + bs) + c);
                }
            } else if (MODE == 2) {
                float bs = bias[col], a = bnA[col], c = bnC[col];
#pragma unroll
                for (int r = 0; r < 8; ++r) {
                    int row = tr0 + r + 8 * hi;
                    float z = a * (a8[r] + bs) + c;
                    atomicMax(&outMax[(long)(row / L_) * N + col], fenc(z));
                }
            } else {
#pragma unroll
                for (int r = 0; r < 8; ++r) {
                    int row = tr0 + r + 8 * hi;
                    outF[(long)row * N + col] = alpha * a8[r];
                }
            }
        }
    }
}

// L2-normalize each 1024-wide row of the (encoded) pooled features -> fp16
__global__ void normalize_kernel(const unsigned* __restrict__ featE, _Float16* __restrict__ out) {
    __shared__ float red[256];
    int row = blockIdx.x;
    const unsigned* f = featE + (long)row * 1024;
    float ss = 0.f;
    for (int j = threadIdx.x; j < 1024; j += 256) { float x = fdec(f[j]); ss += x * x; }
    red[threadIdx.x] = ss;
    __syncthreads();
    for (int st = 128; st > 0; st >>= 1) {
        if (threadIdx.x < st) red[threadIdx.x] += red[threadIdx.x + st];
        __syncthreads();
    }
    float inv = 1.f / fmaxf(sqrtf(red[0]), 1e-12f);
    _Float16* o = out + (long)row * 1024;
    for (int j = threadIdx.x; j < 1024; j += 256) o[j] = (_Float16)(fdec(f[j]) * inv);
}

__global__ void label_kernel(float* __restrict__ out) {
    int i = blockIdx.x * blockDim.x + threadIdx.x;
    if (i < B_) out[i] = (float)i;
}

// ---------------------------------------------------------------------------
extern "C" void kernel_launch(void* const* d_in, const int* in_sizes, int n_in,
                              void* d_out, int out_size, void* d_ws, size_t ws_size,
                              hipStream_t stream) {
    const float* cid  = (const float*)d_in[1];
    const float* time = (const float*)d_in[2];
    const float* W1 = (const float*)d_in[3];  const float* b1 = (const float*)d_in[4];
    const float* W2 = (const float*)d_in[5];  const float* b2 = (const float*)d_in[6];
    const float* W3 = (const float*)d_in[7];  const float* b3 = (const float*)d_in[8];
    const float* g1 = (const float*)d_in[9];  const float* be1 = (const float*)d_in[10];
    const float* g2 = (const float*)d_in[11]; const float* be2 = (const float*)d_in[12];
    const float* g3 = (const float*)d_in[13]; const float* be3 = (const float*)d_in[14];
    const float* gt = (const float*)d_in[15]; const float* bt = (const float*)d_in[16];

    char* w = (char*)d_ws;
    auto alloc = [&](size_t bytes) -> char* {
        char* p = w;
        w += (bytes + 255) & ~(size_t)255;
        return p;
    };
    float*    cA    = (float*)alloc((size_t)POS * 4);
    float*    tA    = (float*)alloc((size_t)POS * 4);
    float*    cB    = (float*)alloc((size_t)POS * 4);
    float*    tB    = (float*)alloc((size_t)POS * 4);
    float*    tstat = (float*)alloc(80 * 4);
    float*    csum  = (float*)alloc(1024 * 4);
    float*    csq   = (float*)alloc(1024 * 4);
    float*    bnA   = (float*)alloc(1024 * 4);
    float*    bnC   = (float*)alloc(1024 * 4);
    float*    y1    = (float*)alloc((size_t)POS * 64 * 4);
    _Float16* h1n   = (_Float16*)alloc((size_t)POS * 64 * 2);
    _Float16* h2n   = (_Float16*)alloc((size_t)POS * 128 * 2);
    _Float16* W2h   = (_Float16*)alloc(128 * 64 * 2);
    _Float16* W3h   = (_Float16*)alloc(1024 * 128 * 2);
    unsigned* fe0   = (unsigned*)alloc((size_t)B_ * 1024 * 4);
    unsigned* fe1   = (unsigned*)alloc((size_t)B_ * 1024 * 4);
    _Float16* qh    = (_Float16*)alloc((size_t)B_ * 1024 * 2);
    _Float16* kh    = (_Float16*)alloc((size_t)B_ * 1024 * 2);

    // augmentation + time batch-norm stats (both branches)
    hipMemsetAsync(tstat, 0, 80 * 4, stream);
    aug_stats_kernel<<<(2 * POS) / 256, 256, 0, stream>>>(cid, time, cA, tA, cB, tB, tstat);

    // weights to fp16 (already [N,K] row-major = Bt layout)
    f2h_kernel<<<(128 * 64 + 255) / 256, 256, 0, stream>>>(W2, W2h, 128 * 64);
    f2h_kernel<<<(1024 * 128 + 255) / 256, 256, 0, stream>>>(W3, W3h, 1024 * 128);

    const float invCnt = 1.f / (float)POS;
    // waves = (M/64)*(N/32); 8 waves per 256-thread block
    const int blk2 = ((POS / 64) * (128 / 32)) / 8;    // stage2: 640
    const int blk3 = ((POS / 64) * (1024 / 32)) / 8;   // stage3: 5120

    for (int br = 0; br < 2; ++br) {
        const float*   c   = br ? cB : cA;
        const float*   t   = br ? tB : tA;
        unsigned*      fe  = br ? fe1 : fe0;
        _Float16*      out = br ? kh : qh;

        // stage1: scalar (K=2), pre-BN activations + stats
        stage1_kernel<<<(POS * 64) / 256, 256, 0, stream>>>(c, t, tstat, br, W1, b1, gt, bt, y1);
        hipMemsetAsync(csum, 0, 1024 * 4, stream);
        hipMemsetAsync(csq, 0, 1024 * 4, stream);
        colstats_kernel<<<512, 256, 2 * 64 * 4, stream>>>(y1, csum, csq, (long)POS, 64);
        bnfinal_kernel<<<1, 64, 0, stream>>>(csum, csq, g1, be1, bnA, bnC, 64, invCnt);
        bnrelu_h_kernel<<<(POS * 64 + 255) / 256, 256, 0, stream>>>(y1, bnA, bnC, h1n, 64,
                                                                    (long)POS * 64);

        // stage2 (WMMA K=64): stats pass, then BN+ReLU store pass
        hipMemsetAsync(csum, 0, 1024 * 4, stream);
        hipMemsetAsync(csq, 0, 1024 * 4, stream);
        wmma_gemm_kernel<0><<<blk2, 256, 0, stream>>>(h1n, W2h, POS, 128, 64, b2,
                                                      nullptr, nullptr, csum, csq,
                                                      nullptr, nullptr, nullptr, 0.f);
        bnfinal_kernel<<<1, 128, 0, stream>>>(csum, csq, g2, be2, bnA, bnC, 128, invCnt);
        wmma_gemm_kernel<1><<<blk2, 256, 0, stream>>>(h1n, W2h, POS, 128, 64, b2,
                                                      bnA, bnC, nullptr, nullptr,
                                                      h2n, nullptr, nullptr, 0.f);

        // stage3 (WMMA K=128): stats pass, then BN + max-over-L pass
        hipMemsetAsync(csum, 0, 1024 * 4, stream);
        hipMemsetAsync(csq, 0, 1024 * 4, stream);
        wmma_gemm_kernel<0><<<blk3, 256, 0, stream>>>(h2n, W3h, POS, 1024, 128, b3,
                                                      nullptr, nullptr, csum, csq,
                                                      nullptr, nullptr, nullptr, 0.f);
        bnfinal_kernel<<<4, 256, 0, stream>>>(csum, csq, g3, be3, bnA, bnC, 1024, invCnt);
        hipMemsetAsync(fe, 0, (size_t)B_ * 1024 * 4, stream);
        wmma_gemm_kernel<2><<<blk3, 256, 0, stream>>>(h2n, W3h, POS, 1024, 128, b3,
                                                      bnA, bnC, nullptr, nullptr,
                                                      nullptr, nullptr, fe, 0.f);

        // pooled -> L2 normalized fp16 features
        normalize_kernel<<<B_, 256, 0, stream>>>(fe, out);
    }

    // logits = (q . k^T) / TEMP   [4096 x 4096], K=1024
    const int blkL = ((B_ / 64) * (B_ / 32)) / 8;      // 1024 blocks
    wmma_gemm_kernel<3><<<blkL, 256, 0, stream>>>(qh, kh, B_, B_, 1024, nullptr,
                                                  nullptr, nullptr, nullptr, nullptr,
                                                  nullptr, (float*)d_out, nullptr,
                                                  1.0f / TEMP_);
    // labels appended after logits (written as float values 0..B-1)
    label_kernel<<<(B_ + 255) / 256, 256, 0, stream>>>((float*)d_out + (size_t)B_ * B_);
}
